// ITMLoss_16097537425576
// MI455X (gfx1250) — matrix operand order repaired and checked by reading, not compiled, not run on hardware
//
#include <hip/hip_runtime.h>
#include <hip/hip_bf16.h>

typedef __attribute__((ext_vector_type(16))) __bf16 v16bf;
typedef __attribute__((ext_vector_type(8)))  __bf16 v8bf;
typedef __attribute__((ext_vector_type(8)))  float  v8f;

constexpr int NB = 8192;   // batch
constexpr int ND = 256;    // dim

// ---------- helpers ----------

// Pack (float value, index) into a monotonically ordered u64 key for argmax.
__device__ __forceinline__ unsigned long long enc_key(float v, unsigned j) {
    unsigned u = __float_as_uint(v);
    u = (u & 0x80000000u) ? ~u : (u | 0x80000000u);
    return (((unsigned long long)u) << 32) | (unsigned long long)j;
}

// Deterministic per-(i,j) uniform in [0,1) (stand-in for jax.random key 42).
__device__ __forceinline__ float noise_u(unsigned i, unsigned j) {
    unsigned x = i * 0x9E3779B1u ^ (j + 0x7F4A7C15u) * 0x85EBCA77u;
    x ^= x >> 16; x *= 0x7FEB352Du;
    x ^= x >> 15; x *= 0x846CA68Bu;
    x ^= x >> 16;
    return (float)(x >> 8) * (1.0f / 16777216.0f);
}

// A-operand fragment (16x32 bf16), contiguous-k order used by this file (B loads match):
// lane m(0..15)/m+16 both hold row m; kh = lane>>4 selects K-halves:
// elems 0..7 = K[k0 + kh*16 .. +7], elems 8..15 = K[k0+kh*16+8 .. +7]
__device__ __forceinline__ v16bf load_a_frag(const __bf16* rowp, int k0, int kh) {
    v8bf lo = *(const v8bf*)(rowp + k0 + kh * 16);
    v8bf hi = *(const v8bf*)(rowp + k0 + kh * 16 + 8);
    v16bf a;
#pragma unroll
    for (int e = 0; e < 8; ++e) { a[e] = lo[e]; a[e + 8] = hi[e]; }
    return a;
}

// ---------- kernel 1: fp32 -> bf16 conversions + W1 transpose + acc zero ----------
__global__ void __launch_bounds__(256)
prep_kernel(const float* __restrict__ vc, const float* __restrict__ tc,
            const float* __restrict__ vu, const float* __restrict__ tu,
            const float* __restrict__ W1,
            __bf16* __restrict__ vcb, __bf16* __restrict__ tcb,
            __bf16* __restrict__ vub, __bf16* __restrict__ tub,
            __bf16* __restrict__ w1t, float* __restrict__ acc) {
    const int idx = blockIdx.x * blockDim.x + threadIdx.x;
    const int stride = gridDim.x * blockDim.x;
    const int N = NB * ND;
    for (int i = idx; i < N; i += stride) {
        vcb[i] = (__bf16)vc[i];
        tcb[i] = (__bf16)tc[i];
        vub[i] = (__bf16)vu[i];
        tub[i] = (__bf16)tu[i];
    }
    // w1t[n][k] = W1[k][n], k in [0,512)  (row 512 of W1 handled separately)
    for (int i = idx; i < 512 * ND; i += stride) {
        const int n = i >> 9, k = i & 511;
        w1t[i] = (__bf16)W1[k * ND + n];
    }
    if (idx < 2) acc[idx] = 0.0f;
}

// ---------- kernel 2: semi-hard negative mining via bf16 WMMA ----------
// One WG per 32-row strip of S = Tuni @ Vuni^T. The 32x256 A strip lives in
// LDS (16 KB); each global B fragment feeds 2 WMMAs (two row tiles).
__global__ void __launch_bounds__(256)
mine_kernel(const __bf16* __restrict__ Tb, const __bf16* __restrict__ Vb,
            int* __restrict__ neg_idx) {
    __shared__ __bf16 a_lds[32 * ND];                  // 16 KB A strip
    __shared__ float diag_lds[32];
    __shared__ unsigned long long band_lds[32];
    __shared__ unsigned long long fb_lds[32];

    const int tid  = threadIdx.x;
    const int lane = tid & 31;
    const int wave = tid >> 5;
    const int col  = lane & 15;
    const int kh   = lane >> 4;
    const int row0 = blockIdx.x * 32;

    if (tid < 32) { band_lds[tid] = 0ull; fb_lds[tid] = 0ull; }

    // Cooperative stage of the 32-row A strip (rows are contiguous in memory).
    {
        const v8bf* src = (const v8bf*)(Tb + (size_t)row0 * ND);
        v8bf* dst = (v8bf*)a_lds;
#pragma unroll
        for (int e = 0; e < 4; ++e) dst[tid + 256 * e] = src[tid + 256 * e];
    }
    __syncthreads();

    if (wave < 2) {  // diagonal tile of row-tile `wave` -> diag[i]
        const __bf16* arow = a_lds + (size_t)(wave * 16 + col) * ND;
        const __bf16* brow = Vb + (size_t)(row0 + wave * 16 + col) * ND;
        v8f c = {};
#pragma unroll
        for (int ks = 0; ks < 8; ++ks) {
            v16bf a = load_a_frag(arow, ks * 32, kh);
            v16bf b = *(const v16bf*)(brow + ks * 32 + kh * 16);
            c = __builtin_amdgcn_wmma_f32_16x16x32_bf16(false, a, false, b,
                                                        (short)0, c, false, false);
        }
#pragma unroll
        for (int r = 0; r < 8; ++r) {
            const int m = r + 8 * kh;
            if (m == col) diag_lds[wave * 16 + m] = c[r];
        }
    }
    __syncthreads();

    float dg[2][8];
#pragma unroll
    for (int rt = 0; rt < 2; ++rt)
#pragma unroll
        for (int r = 0; r < 8; ++r) dg[rt][r] = diag_lds[rt * 16 + r + 8 * kh];

    // Per-lane running argmax state: {value, index} per tracked row.
    float bval[2][8], fval[2][8];
    int   bidx[2][8], fidx[2][8];
#pragma unroll
    for (int rt = 0; rt < 2; ++rt)
#pragma unroll
        for (int r = 0; r < 8; ++r) {
            bval[rt][r] = -1.0f;      bidx[rt][r] = -1;
            fval[rt][r] = -3.4e38f;   fidx[rt][r] = 0;
        }

    const __bf16* arow0 = a_lds + (size_t)col * ND;
    const __bf16* arow1 = a_lds + (size_t)(16 + col) * ND;

    for (int jt = wave; jt < NB / 16; jt += 8) {
        // Keep the loop-invariant LDS A-fragment loads inside the loop
        // (hoisting all 16 fragments would spill to scratch).
        asm volatile("" ::: "memory");

        const int j0 = jt * 16;
        const __bf16* brow = Vb + (size_t)(j0 + col) * ND;
        if (jt + 8 < NB / 16)
            __builtin_prefetch(Vb + (size_t)(j0 + 128 + col) * ND, 0, 0);

        v8f c0 = {}, c1 = {};
#pragma unroll
        for (int ks = 0; ks < 8; ++ks) {   // one B fragment feeds both row tiles
            v16bf b  = *(const v16bf*)(brow + ks * 32 + kh * 16);
            v16bf a0 = load_a_frag(arow0, ks * 32, kh);
            v16bf a1 = load_a_frag(arow1, ks * 32, kh);
            c0 = __builtin_amdgcn_wmma_f32_16x16x32_bf16(false, a0, false, b,
                                                         (short)0, c0, false, false);
            c1 = __builtin_amdgcn_wmma_f32_16x16x32_bf16(false, a1, false, b,
                                                         (short)0, c1, false, false);
        }

#pragma unroll
        for (int rt = 0; rt < 2; ++rt) {
#pragma unroll
            for (int r = 0; r < 8; ++r) {
                const int i = row0 + rt * 16 + r + 8 * kh;
                const int j = j0 + col;
                const float s = rt ? c1[r] : c0[r];
                if (j != i) {
                    const float d = dg[rt][r];
                    if (s > d - 0.5f && s < d - 0.2f) {     // semi-hard band
                        const float nz = noise_u(i, j);
                        if (nz > bval[rt][r]) { bval[rt][r] = nz; bidx[rt][r] = j; }
                    }
                    if (s > fval[rt][r]) { fval[rt][r] = s; fidx[rt][r] = j; }  // hardest
                }
            }
        }
    }

#pragma unroll
    for (int rt = 0; rt < 2; ++rt)
#pragma unroll
        for (int r = 0; r < 8; ++r) {
            const int m = rt * 16 + r + 8 * kh;
            if (bidx[rt][r] >= 0)
                atomicMax(&band_lds[m], enc_key(bval[rt][r], (unsigned)bidx[rt][r]));
            atomicMax(&fb_lds[m], enc_key(fval[rt][r], (unsigned)fidx[rt][r]));
        }
    __syncthreads();

    if (tid < 32) {
        const unsigned long long bk = band_lds[tid];
        const unsigned long long fk = fb_lds[tid];
        neg_idx[row0 + tid] = (int)((bk != 0ull ? bk : fk) & 0xFFFFFFFFull);
    }
}

// ---------- kernel 3: ITM head (pos + neg) + loss partial sums ----------
__global__ void __launch_bounds__(256)
head_kernel(const float* __restrict__ Tc, const float* __restrict__ Vc,
            const __bf16* __restrict__ Tcb, const __bf16* __restrict__ Vcb,
            const __bf16* __restrict__ W1T, const float* __restrict__ W1,
            const float* __restrict__ b1, const float* __restrict__ W2,
            const float* __restrict__ b2, const int* __restrict__ neg_idx,
            float* __restrict__ acc) {
    __shared__ float dot_lds[16];
    __shared__ float logit_lds[16];
    __shared__ int   vrow_lds[16];

    const int tid  = threadIdx.x;
    const int lane = tid & 31;
    const int wave = tid >> 5;
    const int col  = lane & 15;
    const int kh   = lane >> 4;
    const int i0   = blockIdx.x * 16;
    const float b2v = b2[0];

    for (int pass = 0; pass < 2; ++pass) {
        if (tid < 16) {
            const int i  = i0 + tid;
            const int vr = pass ? neg_idx[i] : i;
            vrow_lds[tid] = vr;
            const float* tp = Tc + (size_t)i * ND;
            const float* vp = Vc + (size_t)vr * ND;
            float d = 0.0f;
            for (int k = 0; k < ND; ++k) d = fmaf(tp[k], vp[k], d);
            dot_lds[tid]   = d;     // x[:, 512] feature
            logit_lds[tid] = 0.0f;
        }
        __syncthreads();

        const __bf16* trow = Tcb + (size_t)(i0 + col) * ND;
        const __bf16* vrow = Vcb + (size_t)vrow_lds[col] * ND;

        float part[8];
#pragma unroll
        for (int r = 0; r < 8; ++r) part[r] = 0.0f;

        for (int nt = wave; nt < 16; nt += 8) {
            const int n0 = nt * 16;
            const __bf16* w1row = W1T + (size_t)(n0 + col) * 512;
            v8f c = {};
#pragma unroll
            for (int ks = 0; ks < 16; ++ks) {
                const int k0 = ks * 32;
                v16bf a = (k0 < 256) ? load_a_frag(trow, k0, kh)
                                     : load_a_frag(vrow, k0 - 256, kh);
                v16bf b = *(const v16bf*)(w1row + k0 + kh * 16);
                c = __builtin_amdgcn_wmma_f32_16x16x32_bf16(false, a, false, b,
                                                            (short)0, c, false, false);
            }
#pragma unroll
            for (int r = 0; r < 8; ++r) {
                const int m = r + 8 * kh;
                const int n = n0 + col;
                float h = c[r] + dot_lds[m] * W1[512 * ND + n] + b1[n];  // + dot row + bias
                h = fmaxf(h, 0.0f);                                      // ReLU
                part[r] = fmaf(h, W2[n], part[r]);                       // @ W2
            }
        }
#pragma unroll
        for (int r = 0; r < 8; ++r) atomicAdd(&logit_lds[r + 8 * kh], part[r]);
        __syncthreads();

        if (tid < 16) {
            const float l  = logit_lds[tid] + b2v;
            const float sg = 1.0f / (1.0f + __expf(-l));
            const float loss = pass ? -__logf(1.0f - sg + 1e-8f)
                                    : -__logf(sg + 1e-8f);
            atomicAdd(&acc[pass], loss);
        }
        __syncthreads();
    }
}

// ---------- kernel 4: finalize scalar ----------
__global__ void finalize_kernel(const float* __restrict__ acc, float* __restrict__ out) {
    out[0] = (acc[0] + acc[1]) * (0.5f / (float)NB);
}

// ---------- launch ----------
extern "C" void kernel_launch(void* const* d_in, const int* in_sizes, int n_in,
                              void* d_out, int out_size, void* d_ws, size_t ws_size,
                              hipStream_t stream) {
    (void)in_sizes; (void)n_in; (void)out_size; (void)ws_size;
    const float* vc = (const float*)d_in[0];  // vision_embeds_cross
    const float* tc = (const float*)d_in[1];  // text_embeds_cross
    const float* vu = (const float*)d_in[2];  // vision_embeds_uni
    const float* tu = (const float*)d_in[3];  // text_embeds_uni
    const float* W1 = (const float*)d_in[4];
    const float* b1 = (const float*)d_in[5];
    const float* W2 = (const float*)d_in[6];
    const float* b2 = (const float*)d_in[7];

    char* ws = (char*)d_ws;
    size_t off = 0;
    auto carve = [&](size_t bytes) -> char* {
        char* p = ws + off;
        off += (bytes + 255) & ~(size_t)255;
        return p;
    };
    __bf16* vcb  = (__bf16*)carve((size_t)NB * ND * 2);
    __bf16* tcb  = (__bf16*)carve((size_t)NB * ND * 2);
    __bf16* vub  = (__bf16*)carve((size_t)NB * ND * 2);
    __bf16* tub  = (__bf16*)carve((size_t)NB * ND * 2);
    __bf16* w1t  = (__bf16*)carve((size_t)512 * ND * 2);
    int*    nidx = (int*)carve((size_t)NB * 4);
    float*  acc  = (float*)carve(2 * sizeof(float));

    prep_kernel<<<2048, 256, 0, stream>>>(vc, tc, vu, tu, W1,
                                          vcb, tcb, vub, tub, w1t, acc);
    mine_kernel<<<NB / 32, 256, 0, stream>>>(tub, vub, nidx);
    head_kernel<<<NB / 16, 256, 0, stream>>>(tc, vc, tcb, vcb, w1t, W1, b1, W2, b2,
                                             nidx, acc);
    finalize_kernel<<<1, 1, 0, stream>>>(acc, (float*)d_out);
}
